// GoalConditionedDecoder_48034914238993
// MI455X (gfx1250) — matrix operand order repaired and checked by reading, not compiled, not run on hardware
//
#include <hip/hip_runtime.h>

// GoalConditionedDecoder fused kernel for MI455X (gfx1250, wave32, WMMA).
// B=16384, H=128, G=5, T=12.  One 512-thread workgroup owns BM=32 batch rows,
// stages w_hh as f16 in LDS once, and runs all G*T LSTM steps locally with
// v_wmma_f32_16x16x32_f16.  c stays in registers, h ping-pongs in LDS.
// Position head is split-K parallelized across all threads per step.

#define BB 16384
#define HH 128
#define GG 5
#define TT 12
#define BM 32
#define NTHREADS 512

typedef __attribute__((ext_vector_type(16))) _Float16 v16h;
typedef __attribute__((ext_vector_type(8)))  float    v8f;

// ---- LDS layout (bytes), all 16B aligned ----
#define OFF_WBUF 0         // f16 [128][512] = 131072 (phase1: gp_w1 [128][128] / pr_w1 [128][64])
#define OFF_HB0  131072    // f16 [32][128]  = 8192   (h ping / ghid)
#define OFF_HB1  139264    // f16 [32][128]  = 8192   (h pong / eh16 then phid)
#define OFF_WIH  147456    // f32 [4][512]   = 8192
#define OFF_BIAS 155648    // f32 [512]      = 2048
#define OFF_PHW  157696    // f32 [128][2]   = 1024
#define OFF_GL   158720    // f32 [32][5][2] = 1280
#define OFF_POS  160000    // f32 [32][2]    = 256
#define OFF_LOG  160256    // f32 [32][5]    = 640 (pad)
#define OFF_PHB  160896    // f32 [2]
#define OFF_PACC 160912    // f32 [32][2][8] = 2048 (pos-head split-K partials)
#define SMEM_BYTES 162976

__device__ __forceinline__ float sigmoidf_(float x) {
    return 1.0f / (1.0f + __expf(-x));
}

// A operand (16x32 f16, MxK) from row-major f16 LDS [.., ld] at rows m0..m0+15,
// K base kk.  Per ISA 7.12.2: lane -> M, halves 0..7 -> K=kk+8*half+e,
// halves 8..15 -> K=kk+16+8*half+e.  Two contiguous 16B reads per lane.
__device__ __forceinline__ v16h lds_load_A(const _Float16* base, int ld,
                                           int m0, int kk, int lane) {
    const int half = lane >> 4;
    const _Float16* p = base + (m0 + (lane & 15)) * ld + kk + 8 * half;
    v16h a;
#pragma unroll
    for (int e = 0; e < 8; ++e) { a[e] = p[e]; a[e + 8] = p[e + 16]; }
    return a;
}

// B operand (32x16 f16, KxN) from row-major f16 LDS [.., ld] at K base kk,
// N base n0.  lane -> N (n0+lane%16), halves 0..7 -> K=kk+8*half+e,
// halves 8..15 -> K=kk+16+8*half+e (same K order as A).
__device__ __forceinline__ v16h lds_load_B(const _Float16* base, int ld,
                                           int kk, int n0, int lane) {
    const int half = lane >> 4;
    const _Float16* p = base + (kk + 8 * half) * ld + n0 + (lane & 15);
    v16h b;
#pragma unroll
    for (int e = 0; e < 8; ++e) { b[e] = p[e * ld]; b[e + 8] = p[(e + 16) * ld]; }
    return b;
}

__device__ __forceinline__ v8f wmma_f16(v16h a, v16h b, v8f c) {
    // D = A x B + C, f32 accumulate
    return __builtin_amdgcn_wmma_f32_16x16x32_f16(false, a, false, b,
                                                  (short)0, c, false, false);
}

__global__ __launch_bounds__(NTHREADS)
void gcd_fused_kernel(const float* __restrict__ encoder_hidden,
                      const float* __restrict__ encoder_cell,
                      const float* __restrict__ last_position,
                      const float* __restrict__ gp_w1, const float* __restrict__ gp_b1,
                      const float* __restrict__ gp_w2, const float* __restrict__ gp_b2,
                      const float* __restrict__ pr_w1, const float* __restrict__ pr_b1,
                      const float* __restrict__ pr_w2, const float* __restrict__ pr_b2,
                      const float* __restrict__ w_ih, const float* __restrict__ w_hh,
                      const float* __restrict__ b_ih, const float* __restrict__ b_hh,
                      const float* __restrict__ ph_w, const float* __restrict__ ph_b,
                      float* __restrict__ out) {
    extern __shared__ char smem[];
    _Float16* wbuf   = (_Float16*)(smem + OFF_WBUF);
    _Float16* hb0    = (_Float16*)(smem + OFF_HB0);
    _Float16* hb1    = (_Float16*)(smem + OFF_HB1);
    float*    wih_s  = (float*)(smem + OFF_WIH);
    float*    bias_s = (float*)(smem + OFF_BIAS);
    float*    phw_s  = (float*)(smem + OFF_PHW);
    float*    gl_s   = (float*)(smem + OFF_GL);
    float*    pos_s  = (float*)(smem + OFF_POS);
    float*    log_s  = (float*)(smem + OFF_LOG);
    float*    phb_s  = (float*)(smem + OFF_PHB);
    float*    pacc_s = (float*)(smem + OFF_PACC);

    const int tid  = threadIdx.x;
    const int lane = tid & 31;
    const int wave = tid >> 5;
    const int half = lane >> 4;
    const int nloc = lane & 15;
    const int b0   = blockIdx.x * BM;

    float* out_pred  = out;
    float* out_goals = out + (size_t)BB * GG * TT * 2;
    float* out_probs = out + (size_t)BB * GG * TT * 2 + (size_t)BB * GG * 2;

    // ---------------- stage: eh->f16 (hb1), gp_w1->f16 (wbuf), small consts --
    for (int i = tid; i < BM * HH; i += NTHREADS)
        hb1[i] = (_Float16)encoder_hidden[(size_t)b0 * HH + i];
    for (int i = tid; i < HH * HH; i += NTHREADS)
        wbuf[i] = (_Float16)gp_w1[i];
    for (int i = tid; i < 4 * 4 * HH; i += NTHREADS) wih_s[i] = w_ih[i];
    for (int i = tid; i < 4 * HH; i += NTHREADS) bias_s[i] = b_ih[i] + b_hh[i];
    for (int i = tid; i < HH * 2; i += NTHREADS) phw_s[i] = ph_w[i];
    if (tid < 2) phb_s[tid] = ph_b[tid];
    __syncthreads();

    // ---------------- phase 1a: ghid = relu(eh @ gp_w1 + gp_b1) via WMMA -----
    {
        const int m0 = (wave >> 3) * 16;   // 2 M-tiles
        const int n0 = (wave & 7) * 16;    // 8 N-tiles (H=128)
        v8f acc;
        const float bv = gp_b1[n0 + nloc];
#pragma unroll
        for (int e = 0; e < 8; ++e) acc[e] = bv;
#pragma unroll
        for (int kk = 0; kk < HH; kk += 32) {
            v16h a = lds_load_A(hb1, HH, m0, kk, lane);
            v16h b = lds_load_B(wbuf, HH, kk, n0, lane);
            acc = wmma_f16(a, b, acc);
        }
        __syncthreads();  // all gp_w1 (wbuf) reads done -> safe to restage
#pragma unroll
        for (int e = 0; e < 8; ++e) {
            const int m = m0 + e + 8 * half;
            hb0[m * HH + n0 + nloc] = (_Float16)fmaxf(acc[e], 0.0f);
        }
    }
    for (int i = tid; i < HH * (HH / 2); i += NTHREADS)
        wbuf[i] = (_Float16)pr_w1[i];
    __syncthreads();

    // ---------------- phase 1b: phid = relu(eh @ pr_w1 + pr_b1) (waves 0..7) -
    {
        const bool act = wave < 8;
        const int m0 = (wave >> 2) * 16;   // 2 M-tiles
        const int n0 = (wave & 3) * 16;    // 4 N-tiles (H/2=64)
        v8f acc;
        if (act) {
            const float bv = pr_b1[n0 + nloc];
#pragma unroll
            for (int e = 0; e < 8; ++e) acc[e] = bv;
#pragma unroll
            for (int kk = 0; kk < HH; kk += 32) {
                v16h a = lds_load_A(hb1, HH, m0, kk, lane);
                v16h b = lds_load_B(wbuf, HH / 2, kk, n0, lane);
                acc = wmma_f16(a, b, acc);
            }
        }
        __syncthreads();  // all eh (hb1) reads done -> hb1 reusable for phid
        if (act) {
#pragma unroll
            for (int e = 0; e < 8; ++e) {
                const int m = m0 + e + 8 * half;
                hb1[m * (HH / 2) + n0 + nloc] = (_Float16)fmaxf(acc[e], 0.0f);
            }
        }
        __syncthreads();
    }

    // ---------------- phase 1c: goal heads + logits; stage w_hh -> wbuf ------
    if (tid < BM * GG * 2) {  // 320 threads: goals = ghid @ gp_w2 + b + lastpos
        const int row = tid / (GG * 2);
        const int cc  = tid % (GG * 2);
        float s = gp_b2[cc];
        for (int k = 0; k < HH; ++k)
            s += (float)hb0[row * HH + k] * gp_w2[k * (GG * 2) + cc];
        const float v = s + last_position[(size_t)(b0 + row) * 2 + (cc & 1)];
        gl_s[row * (GG * 2) + cc] = v;
        out_goals[(size_t)(b0 + row) * (GG * 2) + cc] = v;
    }
    if (tid < BM * GG) {      // 160 threads: logits = phid @ pr_w2 + b
        const int row = tid / GG, cc = tid % GG;
        float s = pr_b2[cc];
        for (int k = 0; k < HH / 2; ++k)
            s += (float)hb1[row * (HH / 2) + k] * pr_w2[k * GG + cc];
        log_s[row * GG + cc] = s;
    }
    for (int i = tid; i < HH * 4 * HH; i += NTHREADS)   // 65536 elems -> f16
        wbuf[i] = (_Float16)w_hh[i];
    __syncthreads();
    if (tid < BM) {           // softmax over G=5
        float mx = log_s[tid * GG];
        for (int j = 1; j < GG; ++j) mx = fmaxf(mx, log_s[tid * GG + j]);
        float ex[GG], sum = 0.0f;
        for (int j = 0; j < GG; ++j) { ex[j] = __expf(log_s[tid * GG + j] - mx); sum += ex[j]; }
        const float inv = 1.0f / sum;
        for (int j = 0; j < GG; ++j)
            out_probs[(size_t)(b0 + tid) * GG + j] = ex[j] * inv;
    }

    // ---------------- phase 2: LSTM decoder --------------------------------
    // wave -> (M-tile, J-tile); each wave computes i/f/g/o tiles for its (m,j).
    const int m0 = (wave >> 3) * 16;
    const int j0 = (wave & 7) * 16;
    // pos-head split-K job for this thread: 32 m x 2 col x 8 chunks of 16
    const int pm    = tid >> 4;
    const int pcol  = (tid >> 3) & 1;
    const int pchnk = tid & 7;

    for (int g = 0; g < GG; ++g) {
        for (int i = tid; i < BM * HH; i += NTHREADS)
            hb0[i] = (_Float16)encoder_hidden[(size_t)b0 * HH + i];
        if (tid < BM * 2) pos_s[tid] = last_position[(size_t)b0 * 2 + tid];
        float creg[8];   // c tile in accumulator layout: M=m0+e+8*half, J=j0+nloc
#pragma unroll
        for (int e = 0; e < 8; ++e)
            creg[e] = encoder_cell[(size_t)(b0 + m0 + e + 8 * half) * HH + j0 + nloc];
        __syncthreads();

        int cur = 0;
        for (int t = 0; t < TT; ++t) {
            _Float16* hcur = cur ? hb1 : hb0;
            _Float16* hnxt = cur ? hb0 : hb1;

            // C init: bias + x @ w_ih, x = [pos(2), goal(2)]
            v8f acc[4];
#pragma unroll
            for (int gate = 0; gate < 4; ++gate) {
                const int n = gate * HH + j0 + nloc;
                const float bn = bias_s[n];
                const float w0 = wih_s[0 * 4 * HH + n];
                const float w1 = wih_s[1 * 4 * HH + n];
                const float w2 = wih_s[2 * 4 * HH + n];
                const float w3 = wih_s[3 * 4 * HH + n];
#pragma unroll
                for (int e = 0; e < 8; ++e) {
                    const int m = m0 + e + 8 * half;
                    acc[gate][e] = bn
                        + pos_s[m * 2 + 0] * w0 + pos_s[m * 2 + 1] * w1
                        + gl_s[m * (GG * 2) + g * 2 + 0] * w2
                        + gl_s[m * (GG * 2) + g * 2 + 1] * w3;
                }
            }
            // gates += h @ w_hh  (K=128 -> 4 WMMA per gate, 16 per step per wave)
#pragma unroll
            for (int kk = 0; kk < HH; kk += 32) {
                v16h a = lds_load_A(hcur, HH, m0, kk, lane);
#pragma unroll
                for (int gate = 0; gate < 4; ++gate) {
                    v16h b = lds_load_B(wbuf, 4 * HH, kk, gate * HH + j0, lane);
                    acc[gate] = wmma_f16(a, b, acc[gate]);
                }
            }
            // LSTM nonlinearity, c in registers, h_new -> hnxt (f16)
#pragma unroll
            for (int e = 0; e < 8; ++e) {
                const float iv = acc[0][e], fv = acc[1][e];
                const float gv = acc[2][e], ov = acc[3][e];
                const float cn = sigmoidf_(fv) * creg[e] + sigmoidf_(iv) * tanhf(gv);
                const float hn = sigmoidf_(ov) * tanhf(cn);
                creg[e] = cn;
                const int m = m0 + e + 8 * half;
                hnxt[m * HH + j0 + nloc] = (_Float16)hn;
            }
            __syncthreads();   // hnxt complete; pos/gl reads (above) complete

            // pos head, split-K over all 512 threads: 16 FMAs each
            {
                const _Float16* hr = hnxt + pm * HH + pchnk * 16;
                float s = 0.0f;
#pragma unroll
                for (int k = 0; k < 16; ++k)
                    s += (float)hr[k] * phw_s[(pchnk * 16 + k) * 2 + pcol];
                pacc_s[(pm * 2 + pcol) * 8 + pchnk] = s;
            }
            __syncthreads();
            if (tid < BM * 2) {  // reduce 8 partials; emit prediction
                const int m = tid >> 1, col = tid & 1;
                float s = phb_s[col];
#pragma unroll
                for (int j = 0; j < 8; ++j) s += pacc_s[tid * 8 + j];
                const float pnew = pos_s[tid] + s;
                pos_s[tid] = pnew;
                out_pred[(((size_t)(b0 + m) * GG + g) * TT + t) * 2 + col] = pnew;
            }
            __syncthreads();
            cur ^= 1;
        }
    }
}

extern "C" void kernel_launch(void* const* d_in, const int* in_sizes, int n_in,
                              void* d_out, int out_size, void* d_ws, size_t ws_size,
                              hipStream_t stream) {
    (void)in_sizes; (void)n_in; (void)out_size; (void)d_ws; (void)ws_size;
    const float* encoder_hidden = (const float*)d_in[0];
    const float* encoder_cell   = (const float*)d_in[1];
    const float* last_position  = (const float*)d_in[2];
    const float* gp_w1 = (const float*)d_in[3];
    const float* gp_b1 = (const float*)d_in[4];
    const float* gp_w2 = (const float*)d_in[5];
    const float* gp_b2 = (const float*)d_in[6];
    const float* pr_w1 = (const float*)d_in[7];
    const float* pr_b1 = (const float*)d_in[8];
    const float* pr_w2 = (const float*)d_in[9];
    const float* pr_b2 = (const float*)d_in[10];
    const float* w_ih  = (const float*)d_in[11];
    const float* w_hh  = (const float*)d_in[12];
    const float* b_ih  = (const float*)d_in[13];
    const float* b_hh  = (const float*)d_in[14];
    const float* ph_w  = (const float*)d_in[15];
    const float* ph_b  = (const float*)d_in[16];
    float* outp = (float*)d_out;

    (void)hipFuncSetAttribute((const void*)gcd_fused_kernel,
                              hipFuncAttributeMaxDynamicSharedMemorySize,
                              (int)SMEM_BYTES);
    gcd_fused_kernel<<<BB / BM, NTHREADS, SMEM_BYTES, stream>>>(
        encoder_hidden, encoder_cell, last_position,
        gp_w1, gp_b1, gp_w2, gp_b2,
        pr_w1, pr_b1, pr_w2, pr_b2,
        w_ih, w_hh, b_ih, b_hh, ph_w, ph_b, outp);
}
